// scaledDotProduct_87144886436019
// MI455X (gfx1250) — matrix operand order repaired and checked by reading, not compiled, not run on hardware
//
#include <hip/hip_runtime.h>
#include <math.h>

// ---------------------------------------------------------------------------
// Causal single-head attention forward for MI455X (gfx1250, wave32, WMMA).
//   x:[256,200,1024]f32, Wq/Wk/Wv:[1024,1024]f32 -> out:[256,200,1024]f32
// Pipeline:
//   0) f32->bf16 pre-convert of x and W (halves all GEMM HBM traffic)
//   1) QKV projection, 64x256 tiles, bf16 WMMA, async-to-LDS double buffered
//   2) S = QK^T (WMMA, causal tile skip), 3) softmax, 4) out = P V (WMMA)
// T padded 200 -> 224 (multiple of 32 = bf16 WMMA K-step).
// ---------------------------------------------------------------------------

typedef __bf16 bf16_t;
typedef __attribute__((ext_vector_type(16))) __bf16 v16bf;
typedef __attribute__((ext_vector_type(8)))  float  v8f;

#define LDW 40   // LDS row stride (elems): 80B, multiple of 16B for b128 async

constexpr int    BATCH  = 256;
constexpr int    TSEQ   = 200;
constexpr int    TPAD   = 224;           // 14*16, divisible by 32
constexpr int    CDIM   = 1024;
constexpr int    HDIM   = 1024;
constexpr int    NROWS  = BATCH * TSEQ;  // 51200 (divisible by 64)
constexpr size_t QKV_E  = (size_t)BATCH * TPAD * HDIM;   // bf16 elems / tensor
constexpr size_t S_E    = (size_t)BATCH * TPAD * TPAD;   // f32 elems
constexpr size_t XB_E   = (size_t)NROWS * CDIM;          // bf16 x
constexpr size_t WB_E   = (size_t)3 * HDIM * CDIM;       // bf16 Wq|Wk|Wv

// --- CDNA5 async global->LDS copy (16B per lane, ASYNCcnt tracked) ----------
__device__ __forceinline__ void async_b128(unsigned lds_off, const void* sbase,
                                           unsigned voff) {
  asm volatile("global_load_async_to_lds_b128 %0, %1, %2"
               :: "v"(lds_off), "v"(voff), "s"(sbase) : "memory");
}
__device__ __forceinline__ void wait_async0() {
#if __has_builtin(__builtin_amdgcn_s_wait_asynccnt)
  __builtin_amdgcn_s_wait_asynccnt(0);
#else
  asm volatile("s_wait_asynccnt 0x0" ::: "memory");
#endif
}
__device__ __forceinline__ unsigned lds_off_of(const void* p) {
  return (unsigned)(size_t)p;  // low 32 bits of generic ptr = LDS byte offset
}

// --- WMMA fragment gathers (per CDNA5 ISA 7.12.2 layouts) -------------------
__device__ __forceinline__ v16bf load_frag_a(const bf16_t* lds, int row_base, int lane) {
  const int half = lane >> 4;
  const bf16_t* rowp = lds + (row_base + (lane & 15)) * LDW;
  v16bf a;
#pragma unroll
  for (int j = 0; j < 8; ++j) a[j] = rowp[half * 8 + j];
#pragma unroll
  for (int j = 0; j < 8; ++j) a[8 + j] = rowp[16 + half * 8 + j];
  return a;
}
__device__ __forceinline__ v16bf load_frag_b(const bf16_t* lds, int col_base, int lane) {
  const int half = lane >> 4;
  const bf16_t* colp = lds + (col_base + (lane & 15)) * LDW;
  v16bf b;
#pragma unroll
  for (int j = 0; j < 8; ++j) b[j] = colp[half * 8 + j];
#pragma unroll
  for (int j = 0; j < 8; ++j) b[8 + j] = colp[16 + half * 8 + j];
  return b;
}
#define WMMA_BF16(A, Bm, Cacc) \
  __builtin_amdgcn_wmma_f32_16x16x32_bf16(false, (A), false, (Bm), (short)0, (Cacc), false, false)

// --- Kernel A: convert x and W to bf16 --------------------------------------
__global__ void __launch_bounds__(256) convert_kernel(const float* __restrict__ x,
                                                      const float* __restrict__ Wq,
                                                      const float* __restrict__ Wk,
                                                      const float* __restrict__ Wv,
                                                      bf16_t* __restrict__ xb,
                                                      bf16_t* __restrict__ wb) {
  size_t i = ((size_t)blockIdx.x * 256 + threadIdx.x) * 8;
  const float* src;
  bf16_t* dst;
  if (i < XB_E) {
    src = x + i; dst = xb + i;
  } else {
    size_t e = i - XB_E;
    if (e >= WB_E) return;
    size_t wsel = e / ((size_t)HDIM * CDIM), r = e % ((size_t)HDIM * CDIM);
    src = ((wsel == 0) ? Wq : (wsel == 1) ? Wk : Wv) + r;
    dst = wb + e;
  }
  float4 a = *(const float4*)src;
  float4 b = *(const float4*)(src + 4);
  union { bf16_t h[8]; uint4 u; } t;
  t.h[0] = (bf16_t)a.x; t.h[1] = (bf16_t)a.y; t.h[2] = (bf16_t)a.z; t.h[3] = (bf16_t)a.w;
  t.h[4] = (bf16_t)b.x; t.h[5] = (bf16_t)b.y; t.h[6] = (bf16_t)b.z; t.h[7] = (bf16_t)b.w;
  *(uint4*)dst = t.u;
}

// --- Kernel B: zero pad rows (t in [200,224)) of q,k,v ----------------------
__global__ void __launch_bounds__(256) zero_pad_kernel(bf16_t* __restrict__ qkv) {
  const size_t per = (size_t)BATCH * (TPAD - TSEQ) * HDIM;
  size_t idx = (size_t)blockIdx.x * blockDim.x + threadIdx.x;
  if (idx >= 3 * per) return;
  size_t which = idx / per, rem = idx % per;
  size_t b = rem / ((TPAD - TSEQ) * HDIM);
  size_t r = rem % ((TPAD - TSEQ) * HDIM);
  size_t t = TSEQ + r / HDIM, h = r % HDIM;
  qkv[which * QKV_E + (b * TPAD + t) * HDIM + h] = (bf16_t)0.0f;
}

// --- Kernel 1: QKV projection, 64(M) x 256(N) tile, double-buffered async ---
// grid = (800, 12); blockIdx.y: which = y>>2, h0 = (y&3)*256
__global__ void __launch_bounds__(256) proj_kernel(const bf16_t* __restrict__ xb,
                                                   const bf16_t* __restrict__ wb,
                                                   bf16_t* __restrict__ qkv) {
  __shared__ bf16_t lds_a[2][64 * LDW];
  __shared__ bf16_t lds_b[2][256 * LDW];
  const int tid = threadIdx.x, lane = tid & 31, w = tid >> 5;
  const int wm = w & 3;   // 16-row sub-tile
  const int wn = w >> 2;  // 128-col slab (0/1)
  const int m0 = blockIdx.x * 64;
  const int which = blockIdx.y >> 2;
  const int h0 = (blockIdx.y & 3) * 256;
  const int wrow0 = which * 1024 + h0;
  bf16_t* dst = qkv + (size_t)which * QKV_E;

  v8f acc[8];
#pragma unroll
  for (int t2 = 0; t2 < 8; ++t2) acc[t2] = v8f{};

  const int sr = tid >> 2, sc = (tid & 3) * 8;  // staging row / col-chunk

  auto stage = [&](int buf, int kk) {
    // x: 64x32 tile, 1 chunk/thread
    async_b128(lds_off_of(&lds_a[buf][sr * LDW + sc]), xb,
               (unsigned)((((m0 + sr) << 10) + kk + sc) * 2));
    // W: 256x32 tile, 4 chunks/thread
#pragma unroll
    for (int j = 0; j < 4; ++j) {
      const int r = sr + j * 64;
      async_b128(lds_off_of(&lds_b[buf][r * LDW + sc]), wb,
                 (unsigned)((((wrow0 + r) << 10) + kk + sc) * 2));
    }
  };

  stage(0, 0);
  wait_async0();
  __syncthreads();

  for (int it = 0; it < CDIM / 32; ++it) {
    const int cur = it & 1;
    if (it + 1 < CDIM / 32) stage(cur ^ 1, (it + 1) * 32);
    v16bf a = load_frag_a(lds_a[cur], wm * 16, lane);
#pragma unroll
    for (int t2 = 0; t2 < 8; ++t2) {
      v16bf bm = load_frag_b(lds_b[cur], wn * 128 + t2 * 16, lane);
      acc[t2] = WMMA_BF16(a, bm, acc[t2]);
    }
    wait_async0();
    __syncthreads();
  }

  const int half = lane >> 4, nn = lane & 15;
#pragma unroll
  for (int r = 0; r < 8; ++r) {
    const int mrow = m0 + wm * 16 + r + half * 8;  // < 51200
    const int b = mrow / TSEQ, t = mrow % TSEQ;
    bf16_t* orow = dst + ((size_t)b * TPAD + t) * HDIM;
#pragma unroll
    for (int t2 = 0; t2 < 8; ++t2)
      orow[h0 + wn * 128 + t2 * 16 + nn] = (bf16_t)acc[t2][r];
  }
}

// --- Kernel 2: S = (1/32) * Q K^T, 64x64 tiles, causal skip, async DB -------
// grid = (4, 4, 256)
__global__ void __launch_bounds__(256) scores_kernel(const bf16_t* __restrict__ qb,
                                                     const bf16_t* __restrict__ kb,
                                                     float* __restrict__ S) {
  const int q0 = blockIdx.x * 64, n0 = blockIdx.y * 64, b = blockIdx.z;
  if (n0 > q0 + 63) return;  // fully masked tile
  __shared__ bf16_t lds_a[2][64 * LDW];
  __shared__ bf16_t lds_b[2][64 * LDW];
  const int tid = threadIdx.x, lane = tid & 31, w = tid >> 5;
  const int wm = w & 3, wn = w >> 2;
  const bf16_t* Q = qb + (size_t)b * TPAD * HDIM;
  const bf16_t* K = kb + (size_t)b * TPAD * HDIM;

  v8f acc0 = {}, acc1 = {};
  const int sr = tid >> 2, sc = (tid & 3) * 8;
  int qrow = q0 + sr; if (qrow > TPAD - 1) qrow = TPAD - 1;
  int krow = n0 + sr; if (krow > TPAD - 1) krow = TPAD - 1;

  auto stage = [&](int buf, int kk) {
    async_b128(lds_off_of(&lds_a[buf][sr * LDW + sc]), Q,
               (unsigned)(((qrow << 10) + kk + sc) * 2));
    async_b128(lds_off_of(&lds_b[buf][sr * LDW + sc]), K,
               (unsigned)(((krow << 10) + kk + sc) * 2));
  };

  stage(0, 0);
  wait_async0();
  __syncthreads();

  for (int it = 0; it < HDIM / 32; ++it) {
    const int cur = it & 1;
    if (it + 1 < HDIM / 32) stage(cur ^ 1, (it + 1) * 32);
    v16bf a  = load_frag_a(lds_a[cur], wm * 16, lane);
    v16bf b0 = load_frag_b(lds_b[cur], wn * 32, lane);
    v16bf b1 = load_frag_b(lds_b[cur], wn * 32 + 16, lane);
    acc0 = WMMA_BF16(a, b0, acc0);
    acc1 = WMMA_BF16(a, b1, acc1);
    wait_async0();
    __syncthreads();
  }
  const int half = lane >> 4, nn = lane & 15;
  const float scale = 0.03125f;  // 1024^-0.5
#pragma unroll
  for (int r = 0; r < 8; ++r) {
    const int qt = q0 + wm * 16 + r + half * 8;
    const int kt = n0 + wn * 32 + nn;
    if (qt < TPAD) {
      float* srow = S + ((size_t)b * TPAD + qt) * TPAD;
      if (kt < TPAD) srow[kt] = acc0[r] * scale;
      if (kt + 16 < TPAD) srow[kt + 16] = acc1[r] * scale;
    }
  }
}

// --- Kernel 3: causal softmax row -> P (bf16, masked/pad entries = 0) -------
__global__ void __launch_bounds__(256) softmax_kernel(const float* __restrict__ S,
                                                      bf16_t* __restrict__ P) {
  const int q = blockIdx.x, b = blockIdx.y, tid = threadIdx.x;
  bf16_t* prow = P + ((size_t)b * TPAD + q) * TPAD;
  if (q >= TSEQ) {
    for (int k = tid; k < TPAD; k += 256) prow[k] = (bf16_t)0.0f;
    return;
  }
  const float* srow = S + ((size_t)b * TPAD + q) * TPAD;
  __shared__ float red[256];
  const int kmax = q;

  float lmax = -INFINITY;
  for (int k = tid; k <= kmax; k += 256) lmax = fmaxf(lmax, srow[k]);
  red[tid] = lmax;
  __syncthreads();
  for (int s = 128; s > 0; s >>= 1) {
    if (tid < s) red[tid] = fmaxf(red[tid], red[tid + s]);
    __syncthreads();
  }
  const float rmax = red[0];
  __syncthreads();

  float lsum = 0.0f;
  for (int k = tid; k <= kmax; k += 256) lsum += __expf(srow[k] - rmax);
  red[tid] = lsum;
  __syncthreads();
  for (int s = 128; s > 0; s >>= 1) {
    if (tid < s) red[tid] += red[tid + s];
    __syncthreads();
  }
  const float inv = 1.0f / red[0];

  for (int k = tid; k < TPAD; k += 256) {
    const float p = (k <= kmax) ? __expf(srow[k] - rmax) * inv : 0.0f;
    prow[k] = (bf16_t)p;
  }
}

// --- Kernel 4: out = P V, 64x64 tiles; async P, LDS-transposed V ------------
// grid = (4, 16, 256)
__global__ void __launch_bounds__(256) pv_kernel(const bf16_t* __restrict__ P,
                                                 const bf16_t* __restrict__ V,
                                                 float* __restrict__ out) {
  __shared__ bf16_t lds_a[64 * LDW];
  __shared__ bf16_t lds_b[64 * LDW];
  const int tid = threadIdx.x, lane = tid & 31, w = tid >> 5;
  const int wm = w & 3, wn = w >> 2;
  const int q0 = blockIdx.x * 64, h0 = blockIdx.y * 64, b = blockIdx.z;
  const bf16_t* Pr = P + (size_t)b * TPAD * TPAD;
  const bf16_t* Vr = V + (size_t)b * TPAD * HDIM;

  v8f acc0 = {}, acc1 = {};
  const int arow = tid >> 2, acol = (tid & 3) * 8;  // A: 64x32
  int prow_i = q0 + arow; if (prow_i > TPAD - 1) prow_i = TPAD - 1;
  const int vrow = tid >> 3, vcol = (tid & 7) * 8;  // V: 32(k) x 64(h)

  for (int kk = 0; kk < TPAD; kk += 32) {
    // P tile via async engine
    async_b128(lds_off_of(&lds_a[arow * LDW + acol]), Pr,
               (unsigned)((prow_i * TPAD + kk + acol) * 2));
    // V tile: vector load + transposed LDS store
    union { bf16_t h[8]; uint4 u; } t;
    t.u = *(const uint4*)(Vr + (size_t)(kk + vrow) * HDIM + h0 + vcol);
#pragma unroll
    for (int j = 0; j < 8; ++j) lds_b[(vcol + j) * LDW + vrow] = t.h[j];
    wait_async0();
    __syncthreads();
    v16bf a  = load_frag_a(lds_a, wm * 16, lane);
    v16bf b0 = load_frag_b(lds_b, wn * 32, lane);
    v16bf b1 = load_frag_b(lds_b, wn * 32 + 16, lane);
    acc0 = WMMA_BF16(a, b0, acc0);
    acc1 = WMMA_BF16(a, b1, acc1);
    __syncthreads();
  }
  const int half = lane >> 4, nn = lane & 15;
#pragma unroll
  for (int r = 0; r < 8; ++r) {
    const int qt = q0 + wm * 16 + r + half * 8;
    if (qt < TSEQ) {
      float* orow = out + ((size_t)b * TSEQ + qt) * HDIM;
      const int h = h0 + wn * 32 + nn;
      orow[h] = acc0[r];
      orow[h + 16] = acc1[r];
    }
  }
}

// ---------------------------------------------------------------------------
extern "C" void kernel_launch(void* const* d_in, const int* in_sizes, int n_in,
                              void* d_out, int out_size, void* d_ws, size_t ws_size,
                              hipStream_t stream) {
  const float* x  = (const float*)d_in[0];
  const float* Wq = (const float*)d_in[1];
  const float* Wk = (const float*)d_in[2];
  const float* Wv = (const float*)d_in[3];
  float* out = (float*)d_out;

  // workspace layout (all 16B aligned)
  bf16_t* qb = (bf16_t*)d_ws;
  bf16_t* kb = qb + QKV_E;
  bf16_t* vb = qb + 2 * QKV_E;
  float*  Sb = (float*)(qb + 3 * QKV_E);
  bf16_t* Pb = (bf16_t*)(Sb + S_E);
  bf16_t* xb = Pb + S_E;
  bf16_t* wb = xb + XB_E;

  {  // A) f32 -> bf16 pre-convert (x, Wq|Wk|Wv)
    const size_t tot = (XB_E + WB_E) / 8;
    convert_kernel<<<dim3((unsigned)((tot + 255) / 256)), dim3(256), 0, stream>>>(
        x, Wq, Wk, Wv, xb, wb);
  }
  {  // B) zero pad rows of q,k,v
    const size_t tot = 3ull * BATCH * (TPAD - TSEQ) * HDIM;
    zero_pad_kernel<<<dim3((unsigned)((tot + 255) / 256)), dim3(256), 0, stream>>>(qb);
  }
  proj_kernel<<<dim3(NROWS / 64, 12), dim3(256), 0, stream>>>(xb, wb, qb);
  scores_kernel<<<dim3(4, 4, BATCH), dim3(256), 0, stream>>>(qb, kb, Sb);
  softmax_kernel<<<dim3(TPAD, BATCH), dim3(256), 0, stream>>>(Sb, Pb);
  pv_kernel<<<dim3(4, HDIM / 64, BATCH), dim3(256), 0, stream>>>(Pb, vb, out);
}
